// Wavenet_59657095741925
// MI455X (gfx1250) — matrix operand order repaired and checked by reading, not compiled, not run on hardware
//
#include <hip/hip_runtime.h>
#include <hip/hip_bf16.h>
#include <cstdint>

typedef __attribute__((ext_vector_type(16))) _Float16 v16h;
typedef __attribute__((ext_vector_type(8)))  float    v8f;

#define B_TOTAL   131072
#define N_IN      40
#define D0        128
#define D1        64
#define D2        32
#define N_OUT     33
#define TB        128
#define THREADS   256

// ---------------- workspace layout (bytes) ----------------
#define WSP_OFF   0                          // float [40][128]  (BN-folded trunk W)
#define BSP_OFF   (WSP_OFF + N_IN*D0*4)      // float [128]
#define W1T_OFF   (BSP_OFF + D0*4)           // f16 [33][64][128] (n-major, k contiguous)
#define W2T_OFF   (W1T_OFF + N_OUT*D1*D0*2)  // f16 [33][32][64]
#define WS_BYTES  (W2T_OFF + N_OUT*D2*D1*2)  // ~697 KB

// ---------------- LDS layout (bytes) ----------------
// zt: f16 [128][136] z tile (padded rows = 68 dwords -> conflict-free reads)
// two weight buffers (double-buffered via async-to-LDS):
//   w1t f16[64][136] (17408) | w2t f16[32][72] (4608) | b1 f32[64] | b2 f32[32] | w3 f32[32]
// h1s: per-wave f16 [16][72] re-layout scratch
#define LDS_ZT     0
#define BUF_BYTES  22528
#define BUF_W2T    17408
#define BUF_B1     22016
#define BUF_B2     22272
#define BUF_W3     22400
#define LDS_BUF0   (LDS_ZT + 128*136*2)       // 34816
#define LDS_BUF1   (LDS_BUF0 + BUF_BYTES)     // 57344
#define LDS_H1     (LDS_BUF1 + BUF_BYTES)     // 79872
#define LDS_TOTAL  (LDS_H1 + 8*16*72*2)       // 98304
#define LDS_XT     LDS_BUF0                   // phase-A alias: float [128][40] = 20480B

__device__ __forceinline__ void wave_lds_fence() {
  // intra-wave LDS producer->consumer across lanes; DS ops are in-order per wave.
  asm volatile("s_wait_dscnt 0" ::: "memory");
}

__device__ __forceinline__ void wait_async0() {
  asm volatile("s_wait_asynccnt 0" ::: "memory");
}

// CDNA5 async DMA: global -> LDS, 16B per lane, tracked by ASYNCcnt.
__device__ __forceinline__ void async_b128(uint32_t lds_off, const void* g) {
  asm volatile("global_load_async_to_lds_b128 %0, %1, off"
               :: "v"(lds_off), "v"((uint64_t)(uintptr_t)g)
               : "memory");
}

__device__ __forceinline__ v16h ldfrag(const _Float16* p0, const _Float16* p1) {
  v16h r;
  float4* d = reinterpret_cast<float4*>(&r);
  d[0] = *reinterpret_cast<const float4*>(p0);   // 8 f16 = 16B -> ds_load_b128
  d[1] = *reinterpret_cast<const float4*>(p1);
  return r;
}

__device__ __forceinline__ v8f wmma_f16(v16h a, v16h b, v8f c) {
  return __builtin_amdgcn_wmma_f32_16x16x32_f16(false, a, false, b, (short)0, c,
                                                false, false);
}

// ------------- prep: fold BN into trunk weights -------------
__global__ void prep_trunk(const float* __restrict__ gamma, const float* __restrict__ beta,
                           const float* __restrict__ mean,  const float* __restrict__ var,
                           const float* __restrict__ Ws,    const float* __restrict__ bs,
                           float* __restrict__ wsp, float* __restrict__ bsp) {
  const int n = threadIdx.x;   // 0..127
  float acc = bs[n];
  for (int d = 0; d < N_IN; ++d) {
    const float s = rsqrtf(var[d] + 1e-5f) * gamma[d];
    const float w = Ws[d * D0 + n];
    wsp[d * D0 + n] = w * s;
    acc = fmaf(beta[d] - mean[d] * s, w, acc);
  }
  bsp[n] = acc;
}

// ------------- prep: transpose + f16-convert branch weights -------------
__global__ void prep_conv(const float* __restrict__ W1, const float* __restrict__ W2,
                          _Float16* __restrict__ w1t, _Float16* __restrict__ w2t) {
  const int idx = blockIdx.x * blockDim.x + threadIdx.x;
  const int n1 = N_OUT * D0 * D1;
  if (idx < n1) {
    const int j = idx / (D0 * D1);
    const int rem = idx - j * (D0 * D1);
    const int d = rem >> 6;        // 0..127 (K)
    const int n = rem & 63;        // 0..63  (N)
    w1t[(size_t)j * D1 * D0 + n * D0 + d] = (_Float16)W1[idx];
  } else {
    const int k2 = idx - n1;
    if (k2 < N_OUT * D1 * D2) {
      const int j = k2 / (D1 * D2);
      const int rem = k2 - j * (D1 * D2);
      const int k = rem >> 5;      // 0..63 (K)
      const int m = rem & 31;      // 0..31 (N)
      w2t[(size_t)j * D2 * D1 + m * D1 + k] = (_Float16)W2[k2];
    }
  }
}

// ------------- fused trunk + 33-branch grouped GEMM -------------
__global__ __launch_bounds__(THREADS) void wavenet_fused(
    const float* __restrict__ x, const float* __restrict__ wsp,
    const float* __restrict__ bsp, const uint32_t* __restrict__ w1t_g,
    const uint32_t* __restrict__ w2t_g, const float* __restrict__ b1_g,
    const float* __restrict__ b2_g, const float* __restrict__ w3_g,
    const float* __restrict__ b3_g, float* __restrict__ out) {
  extern __shared__ char smem[];
  _Float16* zt = (_Float16*)(smem + LDS_ZT);
  _Float16* h1s = (_Float16*)(smem + LDS_H1);
  float* xt = (float*)(smem + LDS_XT);

  const int tid = threadIdx.x;
  const int gb0 = blockIdx.x * TB;

  // ---------- phase A: BN-folded trunk 40->128, ReLU, f16 z tile ----------
  {
    const float* xg = x + (size_t)gb0 * N_IN;
    for (int i = tid; i < TB * N_IN; i += THREADS) xt[i] = xg[i];
    __syncthreads();
    const int r  = tid & 127;   // row of tile
    const int hc = tid >> 7;    // column half (uniform per wave -> scalar W reads)
    float xr[N_IN];
#pragma unroll
    for (int d = 0; d < N_IN; ++d) xr[d] = xt[r * N_IN + d];
#pragma unroll 4
    for (int n0 = 0; n0 < 64; ++n0) {
      const int n = hc * 64 + n0;
      float acc = bsp[n];
#pragma unroll
      for (int d = 0; d < N_IN; ++d) acc = fmaf(xr[d], wsp[d * D0 + n], acc);
      zt[r * 136 + n] = (_Float16)fmaxf(acc, 0.f);
    }
  }
  __syncthreads();

  // ---------- phase B: 33 branches, double-buffered async weight staging ----------
  const int wave = tid >> 5;
  const int lane = tid & 31;
  const int ln = lane & 15;
  const int hh = lane >> 4;
  const int rbase = wave * 16;

  // issue async weight DMA for branch j into LDS buffer `bufb`
  auto issue_weights = [&](int j, uint32_t bufb) {
    // W1^T: 1024 x 16B chunks; padded rows of 272B (16B pad per row -> 16B aligned)
    const char* g1 = (const char*)(w1t_g + (size_t)j * (D1 * D0 / 2));
#pragma unroll
    for (int q = 0; q < 4; ++q) {
      const int u = tid + q * 256;                       // 0..1023
      const uint32_t off = bufb + (u >> 4) * 272 + (u & 15) * 16;
      async_b128(off, g1 + u * 16);
    }
    // W2^T: 256 x 16B chunks; padded rows of 144B
    {
      const int u = tid;                                 // 0..255
      const uint32_t off = bufb + BUF_W2T + (u >> 3) * 144 + (u & 7) * 16;
      async_b128(off, (const char*)(w2t_g + (size_t)j * (D2 * D1 / 2)) + u * 16);
    }
    if (tid < 16) async_b128(bufb + BUF_B1 + tid * 16,
                             (const char*)(b1_g + j * D1) + tid * 16);
    if (tid < 8)  async_b128(bufb + BUF_B2 + tid * 16,
                             (const char*)(b2_g + j * D2) + tid * 16);
    if (tid < 8)  async_b128(bufb + BUF_W3 + tid * 16,
                             (const char*)(w3_g + j * D2) + tid * 16);
  };

  // A-fragments of z for this wave's 16-row tile (invariant across branches)
  const _Float16* zrow = zt + (rbase + ln) * 136;
  const v16h a0 = ldfrag(zrow +  0 + 16 * hh, zrow +  8 + 16 * hh);
  const v16h a1 = ldfrag(zrow + 32 + 16 * hh, zrow + 40 + 16 * hh);
  const v16h a2 = ldfrag(zrow + 64 + 16 * hh, zrow + 72 + 16 * hh);
  const v16h a3 = ldfrag(zrow + 96 + 16 * hh, zrow + 104 + 16 * hh);

  _Float16* h1w = h1s + wave * 16 * 72;

  issue_weights(0, LDS_BUF0);   // prologue (xt alias dead after the barrier above)

  for (int j = 0; j < N_OUT; ++j) {
    const uint32_t bufb = (j & 1) ? LDS_BUF1 : LDS_BUF0;
    // my async DMAs done -> barrier -> everyone's DMAs done & prev compute finished
    wait_async0();
    __syncthreads();
    if (j + 1 < N_OUT) issue_weights(j + 1, (j & 1) ? LDS_BUF0 : LDS_BUF1);

    const _Float16* w1t = (const _Float16*)(smem + bufb);
    const _Float16* w2t = (const _Float16*)(smem + bufb + BUF_W2T);
    const float* b1s = (const float*)(smem + bufb + BUF_B1);
    const float* b2s = (const float*)(smem + bufb + BUF_B2);
    const float* w3s = (const float*)(smem + bufb + BUF_W3);

    // ---- stage 1: h1 = ReLU(z @ W1_j + b1)   [16x128]x[128x64] ----
    // two interleaved accumulator chains per pass: independent WMMAs pipeline,
    // LDS fragment loads overlap with the other chain's matrix op.
#pragma unroll
    for (int ntp = 0; ntp < 2; ++ntp) {
      const int nt0 = 2 * ntp, nt1 = 2 * ntp + 1;
      const float bb0 = b1s[16 * nt0 + ln];
      const float bb1 = b1s[16 * nt1 + ln];
      v8f acc0 = {bb0, bb0, bb0, bb0, bb0, bb0, bb0, bb0};
      v8f acc1 = {bb1, bb1, bb1, bb1, bb1, bb1, bb1, bb1};
      const _Float16* br0 = w1t + (16 * nt0 + ln) * 136;
      const _Float16* br1 = w1t + (16 * nt1 + ln) * 136;

      v16h p0 = ldfrag(br0 +  0 + 16 * hh, br0 +  8 + 16 * hh);
      v16h q0 = ldfrag(br1 +  0 + 16 * hh, br1 +  8 + 16 * hh);
      v16h p1 = ldfrag(br0 + 32 + 16 * hh, br0 + 40 + 16 * hh);
      v16h q1 = ldfrag(br1 + 32 + 16 * hh, br1 + 40 + 16 * hh);
      acc0 = wmma_f16(a0, p0, acc0);
      acc1 = wmma_f16(a0, q0, acc1);
      v16h p2 = ldfrag(br0 + 64 + 16 * hh, br0 + 72 + 16 * hh);
      v16h q2 = ldfrag(br1 + 64 + 16 * hh, br1 + 72 + 16 * hh);
      acc0 = wmma_f16(a1, p1, acc0);
      acc1 = wmma_f16(a1, q1, acc1);
      v16h p3 = ldfrag(br0 + 96 + 16 * hh, br0 + 104 + 16 * hh);
      v16h q3 = ldfrag(br1 + 96 + 16 * hh, br1 + 104 + 16 * hh);
      acc0 = wmma_f16(a2, p2, acc0);
      acc1 = wmma_f16(a2, q2, acc1);
      acc0 = wmma_f16(a3, p3, acc0);
      acc1 = wmma_f16(a3, q3, acc1);

#pragma unroll
      for (int v = 0; v < 8; ++v) {
        const float t0 = fmaxf(acc0[v], 0.f);
        const float t1 = fmaxf(acc1[v], 0.f);
        h1w[(v + 8 * hh) * 72 + 16 * nt0 + ln] = (_Float16)t0;  // C-layout -> row-major
        h1w[(v + 8 * hh) * 72 + 16 * nt1 + ln] = (_Float16)t1;
      }
    }
    wave_lds_fence();

    // ---- stage 2: h2 = ReLU(h1 @ W2_j + b2)   [16x64]x[64x32], head in-register ----
    const _Float16* h1row = h1w + ln * 72;
    const v16h c0 = ldfrag(h1row +  0 + 16 * hh, h1row +  8 + 16 * hh);
    const v16h c1 = ldfrag(h1row + 32 + 16 * hh, h1row + 40 + 16 * hh);

    const _Float16* br0 = w2t + ln * 72;          // nt = 0 row
    const _Float16* br1 = w2t + (16 + ln) * 72;   // nt = 1 row
    const v16h e00 = ldfrag(br0 +  0 + 16 * hh, br0 +  8 + 16 * hh);
    const v16h e01 = ldfrag(br0 + 32 + 16 * hh, br0 + 40 + 16 * hh);
    const v16h e10 = ldfrag(br1 +  0 + 16 * hh, br1 +  8 + 16 * hh);
    const v16h e11 = ldfrag(br1 + 32 + 16 * hh, br1 + 40 + 16 * hh);

    const float bb0 = b2s[ln], bb1 = b2s[16 + ln];
    v8f acc0 = {bb0, bb0, bb0, bb0, bb0, bb0, bb0, bb0};
    v8f acc1 = {bb1, bb1, bb1, bb1, bb1, bb1, bb1, bb1};
    acc0 = wmma_f16(c0, e00, acc0);
    acc1 = wmma_f16(c0, e10, acc1);
    acc0 = wmma_f16(c1, e01, acc0);
    acc1 = wmma_f16(c1, e11, acc1);

    // ---- stage 3: gu[m] = sum_n h2[m][n]*W3[n] + b3  (butterfly over 16-lane half) ----
    const float w3v0 = w3s[ln], w3v1 = w3s[16 + ln];
    float partial[8];
#pragma unroll
    for (int v = 0; v < 8; ++v)
      partial[v] = fmaxf(acc0[v], 0.f) * w3v0 + fmaxf(acc1[v], 0.f) * w3v1;

#pragma unroll
    for (int v = 0; v < 8; ++v) {
      float p = partial[v];
      p += __shfl_xor(p, 1, 32);
      p += __shfl_xor(p, 2, 32);
      p += __shfl_xor(p, 4, 32);
      p += __shfl_xor(p, 8, 32);
      partial[v] = p;   // gu for row (v + 8*hh), replicated across the half
    }
    const float b3j = b3_g[j];
    if (ln == 0) {
#pragma unroll
      for (int v = 0; v < 8; ++v) {
        const int row = gb0 + rbase + v + 8 * hh;
        out[(size_t)row * N_OUT + j] = partial[v] + b3j;
      }
    }
  }
}

extern "C" void kernel_launch(void* const* d_in, const int* in_sizes, int n_in,
                              void* d_out, int out_size, void* d_ws, size_t ws_size,
                              hipStream_t stream) {
  const float* x     = (const float*)d_in[0];
  const float* gamma = (const float*)d_in[1];
  const float* beta  = (const float*)d_in[2];
  const float* mean  = (const float*)d_in[3];
  const float* var   = (const float*)d_in[4];
  const float* Ws    = (const float*)d_in[5];
  const float* bs    = (const float*)d_in[6];
  const float* W1    = (const float*)d_in[7];
  const float* b1    = (const float*)d_in[8];
  const float* W2    = (const float*)d_in[9];
  const float* b2    = (const float*)d_in[10];
  const float* W3    = (const float*)d_in[11];
  const float* b3    = (const float*)d_in[12];
  float* out = (float*)d_out;

  char* ws = (char*)d_ws;
  float*    wsp = (float*)(ws + WSP_OFF);
  float*    bsp = (float*)(ws + BSP_OFF);
  _Float16* w1t = (_Float16*)(ws + W1T_OFF);
  _Float16* w2t = (_Float16*)(ws + W2T_OFF);

  (void)in_sizes; (void)n_in; (void)out_size; (void)ws_size;

  hipFuncSetAttribute((const void*)wavenet_fused,
                      hipFuncAttributeMaxDynamicSharedMemorySize, LDS_TOTAL);

  prep_trunk<<<1, 128, 0, stream>>>(gamma, beta, mean, var, Ws, bs, wsp, bsp);
  const int totalc = N_OUT * D0 * D1 + N_OUT * D1 * D2;
  prep_conv<<<(totalc + 255) / 256, 256, 0, stream>>>(W1, W2, w1t, w2t);
  wavenet_fused<<<B_TOTAL / TB, THREADS, LDS_TOTAL, stream>>>(
      x, wsp, bsp, (const uint32_t*)w1t, (const uint32_t*)w2t,
      b1, b2, W3, b3, out);
}
